// MultiHeadSelfAttention_5497558139385
// MI455X (gfx1250) — compile-verified, matrix operand order checked by reading, not run on hardware
//
#include <hip/hip_runtime.h>
#include <hip/hip_bf16.h>
#include <stdint.h>

#define DM 1024
#define NH 16
#define DH 64
#define BB 2
#define TT 2048

typedef __attribute__((ext_vector_type(16))) __bf16 v16bf;
typedef __attribute__((ext_vector_type(8)))  float  v8f;

union FragU { v16bf v; uint4 q[2]; };

__device__ __forceinline__ uint16_t f2bf(float f) {
    uint32_t u = __float_as_uint(f);
    u += 0x7FFFu + ((u >> 16) & 1u);   // round-to-nearest-even
    return (uint16_t)(u >> 16);
}

// Load one 16-bit WMMA A/B fragment (16x32 elements, K-chunk starting at kbase)
// from a row-major source. rowptr = start of this lane's row (row = M or N = lane%16).
// Per ISA layout: lane half hb = lane/16; elements 0..7 -> K = kbase+8*hb+0..7,
// elements 8..15 -> K = kbase+16+8*hb+0..7. Both are contiguous 16-byte chunks.
__device__ __forceinline__ v16bf load_frag_g(const uint16_t* rowptr, int kbase, int hb) {
    FragU f;
    f.q[0] = *(const uint4*)(rowptr + kbase + 8 * hb);
    f.q[1] = *(const uint4*)(rowptr + kbase + 16 + 8 * hb);
    return f.v;
}

__device__ __forceinline__ v8f wmma_bf16(v16bf a, v16bf b, v8f c) {
    return __builtin_amdgcn_wmma_f32_16x16x32_bf16(false, a, false, b, (short)0, c, false, false);
}

// ---------------- conversion kernels ----------------

__global__ void cvt_x(const float* __restrict__ x, uint16_t* __restrict__ xb) {
    int i = blockIdx.x * blockDim.x + threadIdx.x;
    xb[i] = f2bf(x[i]);
}

// Build W_qkv^T bf16: row n in [0,3072), col d in [0,1024). n -> (which,h,f).
__global__ void cvt_wqkv(const float* __restrict__ Wq, const float* __restrict__ Wk,
                         const float* __restrict__ Wv, uint16_t* __restrict__ wT) {
    int i = blockIdx.x * blockDim.x + threadIdx.x;   // over 3*1024*1024
    int ncol = i >> 10;
    int d    = i & 1023;
    int which = ncol >> 10;
    int hf    = ncol & 1023;
    int h = hf >> 6, f = hf & 63;
    const float* W = (which == 0) ? Wq : ((which == 1) ? Wk : Wv);
    wT[i] = f2bf(W[((size_t)h * DM + d) * DH + f]);
}

// Build W_O^T bf16: row d, col hf.  woT[d,hf] = W_O[h,f,d]
__global__ void cvt_wo(const float* __restrict__ Wo, uint16_t* __restrict__ wT) {
    int i = blockIdx.x * blockDim.x + threadIdx.x;   // over 1024*1024
    int d = i >> 10, hf = i & 1023;
    int h = hf >> 6, f = hf & 63;
    wT[i] = f2bf(Wo[((size_t)(h * DH + f)) * DM + d]);
}

__global__ void sum_bo(const float* __restrict__ bo, float* __restrict__ bs) {
    int d = blockIdx.x * blockDim.x + threadIdx.x;
    float s = 0.f;
    #pragma unroll
    for (int h = 0; h < NH; h++) s += bo[h * DM + d];
    bs[d] = s;
}

// ---------------- QKV projection GEMM (32x32 tile per wave) ----------------
// C[t, n] = sum_d xb[t,d] * wT[n,d] + bias(n)
__global__ void qkv_gemm(const uint16_t* __restrict__ xb, const uint16_t* __restrict__ wT,
                         const float* __restrict__ bq, const float* __restrict__ bk,
                         const float* __restrict__ bv,
                         uint16_t* __restrict__ qb, uint16_t* __restrict__ kbuf,
                         uint16_t* __restrict__ vtb) {
    const int lane  = threadIdx.x & 31;
    const int hb    = lane >> 4;
    const int lanem = lane & 15;
    const int wave  = blockIdx.x * (blockDim.x >> 5) + (threadIdx.x >> 5);
    const int NT = (3 * DM) / 32;          // 96 column tiles
    const int t0 = (wave / NT) * 32;
    const int n0 = (wave % NT) * 32;

    const uint16_t* arow0 = xb + (size_t)(t0 + lanem) * DM;
    const uint16_t* arow1 = arow0 + 16 * DM;
    const uint16_t* brow0 = wT + (size_t)(n0 + lanem) * DM;
    const uint16_t* brow1 = brow0 + 16 * DM;

    v8f c00 = {}, c01 = {}, c10 = {}, c11 = {};
    for (int kk = 0; kk < DM; kk += 32) {
        __builtin_prefetch(arow0 + kk + 64, 0, 1);
        __builtin_prefetch(arow1 + kk + 64, 0, 1);
        __builtin_prefetch(brow0 + kk + 64, 0, 1);
        __builtin_prefetch(brow1 + kk + 64, 0, 1);
        v16bf a0 = load_frag_g(arow0, kk, hb);
        v16bf a1 = load_frag_g(arow1, kk, hb);
        v16bf b0 = load_frag_g(brow0, kk, hb);
        v16bf b1 = load_frag_g(brow1, kk, hb);
        c00 = wmma_bf16(a0, b0, c00);
        c01 = wmma_bf16(a0, b1, c01);
        c10 = wmma_bf16(a1, b0, c10);
        c11 = wmma_bf16(a1, b1, c11);
    }

    #pragma unroll
    for (int nj = 0; nj < 2; nj++) {
        const int ncol  = n0 + nj * 16 + lanem;
        const int which = ncol >> 10;      // wave-uniform per sub-tile
        const int hf    = ncol & 1023;
        const int h = hf >> 6, f = hf & 63;
        const float bias = (which == 0) ? bq[hf] : ((which == 1) ? bk[hf] : bv[hf]);
        #pragma unroll
        for (int mi = 0; mi < 2; mi++) {
            const v8f& c = (mi == 0) ? (nj == 0 ? c00 : c01) : (nj == 0 ? c10 : c11);
            #pragma unroll
            for (int r = 0; r < 8; r++) {
                int t  = t0 + mi * 16 + r + 8 * hb;
                int b_ = t >> 11;          // / 2048
                int tt = t & 2047;
                uint16_t val = f2bf(c[r] + bias);
                if (which == 0)      qb  [(((size_t)(b_ * NH + h)) * TT + tt) * DH + f] = val;
                else if (which == 1) kbuf[(((size_t)(b_ * NH + h)) * TT + tt) * DH + f] = val;
                else                 vtb [(((size_t)(b_ * NH + h)) * DH + f) * TT + tt] = val;
            }
        }
    }
}

// ---------------- flash attention ----------------
// One wave handles 16 query rows of one (b,h); streams 32 keys per step.
__global__ void attn_kernel(const uint16_t* __restrict__ qb, const uint16_t* __restrict__ kb,
                            const uint16_t* __restrict__ vtb, const float* __restrict__ mask,
                            uint16_t* __restrict__ ctx) {
    __shared__ uint16_t plds[4 * 16 * 32];           // 1KB per wave (P spill)
    const int lane  = threadIdx.x & 31;
    const int hb    = lane >> 4;
    const int lanem = lane & 15;
    const int wsl   = threadIdx.x >> 5;
    const int wave  = blockIdx.x * 4 + wsl;
    const int row_blk = wave & 127;                  // T/16 = 128 blocks
    const int bh = wave >> 7;                        // 0..31
    const int h = bh & 15, b = bh >> 4;
    const int m0 = row_blk * 16;
    uint16_t* pl = plds + wsl * 512;

    // Q fragments for this 16-row block (kept in registers for whole loop)
    const uint16_t* qrow = qb + ((size_t)bh * TT + m0 + lanem) * DH;
    const v16bf qf0 = load_frag_g(qrow, 0,  hb);
    const v16bf qf1 = load_frag_g(qrow, 32, hb);

    const uint16_t* kbase = kb + (size_t)bh * TT * DH;
    const uint16_t* vrow[4];
    #pragma unroll
    for (int ft = 0; ft < 4; ft++)
        vrow[ft] = vtb + ((size_t)bh * DH + ft * 16 + lanem) * TT;

    float m_run[8], l_run[8];
    v8f acc[4] = {};
    #pragma unroll
    for (int r = 0; r < 8; r++) { m_run[r] = -1e30f; l_run[r] = 0.f; }

    const float* maskrow0 = mask + (size_t)m0 * TT;

    for (int s0 = 0; s0 < TT; s0 += 32) {
        // ---- S = Q K^T for 32 columns (two 16x16 tiles, K=64 over F) ----
        const uint16_t* krow  = kbase + (size_t)(s0 + lanem) * DH;
        const uint16_t* krow2 = krow + 16 * DH;
        v16bf k0a = load_frag_g(krow,  0,  hb);
        v16bf k0b = load_frag_g(krow,  32, hb);
        v16bf k1a = load_frag_g(krow2, 0,  hb);
        v16bf k1b = load_frag_g(krow2, 32, hb);
        v8f s_a = {}, s_b = {};
        s_a = wmma_bf16(qf0, k0a, s_a);
        s_a = wmma_bf16(qf1, k0b, s_a);
        s_b = wmma_bf16(qf0, k1a, s_b);
        s_b = wmma_bf16(qf1, k1b, s_b);

        // scale (1/sqrt(64)) + additive mask
        #pragma unroll
        for (int r = 0; r < 8; r++) {
            const float* mr = maskrow0 + (size_t)(r + 8 * hb) * TT + s0 + lanem;
            s_a[r] = s_a[r] * 0.125f + mr[0];
            s_b[r] = s_b[r] * 0.125f + mr[16];
        }

        // ---- online softmax update ----
        #pragma unroll
        for (int r = 0; r < 8; r++) {
            float rowmax = fmaxf(s_a[r], s_b[r]);
            #pragma unroll
            for (int off = 1; off < 16; off <<= 1)
                rowmax = fmaxf(rowmax, __shfl_xor(rowmax, off, 32));
            float mnew  = fmaxf(m_run[r], rowmax);
            float alpha = __expf(m_run[r] - mnew);
            m_run[r] = mnew;
            float p0 = __expf(s_a[r] - mnew);
            float p1 = __expf(s_b[r] - mnew);
            s_a[r] = p0; s_b[r] = p1;
            float rs = p0 + p1;
            #pragma unroll
            for (int off = 1; off < 16; off <<= 1)
                rs += __shfl_xor(rs, off, 32);
            l_run[r] = l_run[r] * alpha + rs;
            #pragma unroll
            for (int ft = 0; ft < 4; ft++) acc[ft][r] *= alpha;
        }

        // ---- C-layout -> A-layout conversion of P through LDS (bf16) ----
        #pragma unroll
        for (int r = 0; r < 8; r++) {
            int row = r + 8 * hb;
            pl[row * 32 + lanem]      = f2bf(s_a[r]);
            pl[row * 32 + 16 + lanem] = f2bf(s_b[r]);
        }
        asm volatile("s_wait_dscnt 0x0" ::: "memory");   // cross-lane LDS RAW (same wave)
        FragU pf;
        const uint16_t* prow = pl + lanem * 32;
        pf.q[0] = *(const uint4*)(prow + 8 * hb);
        pf.q[1] = *(const uint4*)(prow + 16 + 8 * hb);

        // ---- O += P V  (4 f-tiles of 16, K=32 over s) ----
        #pragma unroll
        for (int ft = 0; ft < 4; ft++) {
            v16bf vf = load_frag_g(vrow[ft], s0, hb);
            acc[ft] = wmma_bf16(pf.v, vf, acc[ft]);
        }
    }

    // normalize and write context as bf16 [B*T, H*F]
    #pragma unroll
    for (int r = 0; r < 8; r++) {
        float inv = 1.0f / l_run[r];
        int trow = b * TT + m0 + r + 8 * hb;
        size_t base = (size_t)trow * DM + h * DH;
        #pragma unroll
        for (int ft = 0; ft < 4; ft++)
            ctx[base + ft * 16 + lanem] = f2bf(acc[ft][r] * inv);
    }
}

// ---------------- output projection GEMM (32x32 tile per wave) ----------------
__global__ void out_gemm(const uint16_t* __restrict__ ctxb, const uint16_t* __restrict__ woT,
                         const float* __restrict__ bsum, float* __restrict__ out) {
    const int lane  = threadIdx.x & 31;
    const int hb    = lane >> 4;
    const int lanem = lane & 15;
    const int wave  = blockIdx.x * (blockDim.x >> 5) + (threadIdx.x >> 5);
    const int NT = DM / 32;                // 32 column tiles
    const int t0 = (wave / NT) * 32;
    const int n0 = (wave % NT) * 32;

    const uint16_t* arow0 = ctxb + (size_t)(t0 + lanem) * DM;
    const uint16_t* arow1 = arow0 + 16 * DM;
    const uint16_t* brow0 = woT + (size_t)(n0 + lanem) * DM;
    const uint16_t* brow1 = brow0 + 16 * DM;

    v8f c00 = {}, c01 = {}, c10 = {}, c11 = {};
    for (int kk = 0; kk < DM; kk += 32) {
        __builtin_prefetch(arow0 + kk + 64, 0, 1);
        __builtin_prefetch(arow1 + kk + 64, 0, 1);
        __builtin_prefetch(brow0 + kk + 64, 0, 1);
        __builtin_prefetch(brow1 + kk + 64, 0, 1);
        v16bf a0 = load_frag_g(arow0, kk, hb);
        v16bf a1 = load_frag_g(arow1, kk, hb);
        v16bf b0 = load_frag_g(brow0, kk, hb);
        v16bf b1 = load_frag_g(brow1, kk, hb);
        c00 = wmma_bf16(a0, b0, c00);
        c01 = wmma_bf16(a0, b1, c01);
        c10 = wmma_bf16(a1, b0, c10);
        c11 = wmma_bf16(a1, b1, c11);
    }

    #pragma unroll
    for (int nj = 0; nj < 2; nj++) {
        const int d = n0 + nj * 16 + lanem;
        const float bias = bsum[d];
        #pragma unroll
        for (int mi = 0; mi < 2; mi++) {
            const v8f& c = (mi == 0) ? (nj == 0 ? c00 : c01) : (nj == 0 ? c10 : c11);
            #pragma unroll
            for (int r = 0; r < 8; r++)
                out[(size_t)(t0 + mi * 16 + r + 8 * hb) * DM + d] = c[r] + bias;
        }
    }
}

// ---------------- host launcher ----------------
extern "C" void kernel_launch(void* const* d_in, const int* in_sizes, int n_in,
                              void* d_out, int out_size, void* d_ws, size_t ws_size,
                              hipStream_t stream) {
    const float* x    = (const float*)d_in[0];
    const float* mask = (const float*)d_in[1];
    const float* Wq   = (const float*)d_in[2];
    const float* Wk   = (const float*)d_in[3];
    const float* Wv   = (const float*)d_in[4];
    const float* Wo   = (const float*)d_in[5];
    const float* bq   = (const float*)d_in[6];
    const float* bk   = (const float*)d_in[7];
    const float* bv   = (const float*)d_in[8];
    const float* bo   = (const float*)d_in[9];
    float* out = (float*)d_out;

    char* ws = (char*)d_ws;
    const size_t MB = 1u << 20;
    uint16_t* xb    = (uint16_t*)(ws);             //  8 MB: x bf16 [4096,1024]
    uint16_t* wqkvT = (uint16_t*)(ws + 8 * MB);    //  6 MB: W_qkv^T bf16 [3072,1024]
    uint16_t* woT   = (uint16_t*)(ws + 14 * MB);   //  2 MB: W_O^T bf16 [1024,1024]
    uint16_t* qb    = (uint16_t*)(ws + 16 * MB);   //  8 MB: Q bf16 [B,H,T,F]
    uint16_t* kbuf  = (uint16_t*)(ws + 24 * MB);   //  8 MB: K bf16 [B,H,T,F]
    uint16_t* vtb   = (uint16_t*)(ws + 32 * MB);   //  8 MB: V^T bf16 [B,H,F,T]
    uint16_t* ctx   = (uint16_t*)(ws + 40 * MB);   //  8 MB: context bf16 [B*T, H*F]
    float*    bsum  = (float*)(ws + 48 * MB);      //  4 KB: sum_h b_O

    cvt_x   <<<(BB * TT * DM) / 256, 256, 0, stream>>>(x, xb);
    cvt_wqkv<<<(3 * DM * DM) / 256, 256, 0, stream>>>(Wq, Wk, Wv, wqkvT);
    cvt_wo  <<<(DM * DM) / 256, 256, 0, stream>>>(Wo, woT);
    sum_bo  <<<DM / 256, 256, 0, stream>>>(bo, bsum);

    // (BT/32)*(3D/32) = 128*96 = 12288 tiles, 8 waves per 256-thread block
    qkv_gemm<<<12288 / 8, 256, 0, stream>>>(xb, wqkvT, bq, bk, bv, qb, kbuf, vtb);

    // B*H*(T/16) = 4096 row-blocks, 4 waves per 128-thread block
    attn_kernel<<<4096 / 4, 128, 0, stream>>>(qb, kbuf, vtb, mask, ctx);

    // (BT/32)*(D/32) = 128*32 = 4096 tiles
    out_gemm<<<4096 / 8, 256, 0, stream>>>(ctx, woT, bsum, out);
}
